// GemmaAttention_49855980372242
// MI455X (gfx1250) — matrix operand order repaired and checked by reading, not compiled, not run on hardware
//
#include <hip/hip_runtime.h>
#include <hip/hip_bf16.h>

// ---------------------------------------------------------------------------
// Gemma attention for MI455X (gfx1250, wave32, WMMA 16x16x32 f16 -> f32 acc)
// ---------------------------------------------------------------------------

typedef _Float16 h8_t  __attribute__((ext_vector_type(8)));
typedef _Float16 h16_t __attribute__((ext_vector_type(16)));
typedef float    f8_t  __attribute__((ext_vector_type(8)));

#define B_    2
#define S_    2048
#define HID_  3072
#define H_    16
#define HKV_  8
#define D_    256

static __device__ __forceinline__ h16_t cat8(h8_t lo, h8_t hi) {
  return __builtin_shufflevector(lo, hi, 0,1,2,3,4,5,6,7,8,9,10,11,12,13,14,15);
}

static __device__ __forceinline__ h8_t cvt8(float4 a, float4 b) {
  h8_t r;
  r[0] = (_Float16)a.x; r[1] = (_Float16)a.y; r[2] = (_Float16)a.z; r[3] = (_Float16)a.w;
  r[4] = (_Float16)b.x; r[5] = (_Float16)b.y; r[6] = (_Float16)b.z; r[7] = (_Float16)b.w;
  return r;
}

static __device__ __forceinline__ f8_t wmma16(h16_t a, h16_t b, f8_t c) {
  // D = A(16x32 f16) x B(32x16 f16) + C(16x16 f32)
  return __builtin_amdgcn_wmma_f32_16x16x32_f16(false, a, false, b, (short)0, c,
                                                false, false);
}

// ---------------------------------------------------------------------------
// Generic tiled WMMA GEMM: C[M,N] = A[M,K] * B[K,N]
//   A: fp32 (converted on the fly) or f16; B: fp32 weights; C: f16 or fp32
//   BM=128, BN=128, BK=32; 256 threads = 8 waves; wave tile = 32x64
// ---------------------------------------------------------------------------
template <bool A16, bool OUT32>
__global__ __launch_bounds__(256) void gemm_wmma_kernel(
    const void* __restrict__ Ap, const float* __restrict__ Bp,
    void* __restrict__ Cp, int M, int N, int K)
{
  __shared__ _Float16 As[128][40];   // [m][k] pad->40 halves (80B, 16B aligned)
  __shared__ _Float16 Bt[128][40];   // [n][k] transposed for contiguous B-frags

  const int tid  = threadIdx.x;
  const int lane = tid & 31;
  const int wid  = tid >> 5;
  const int cl   = lane & 15;
  const int hi   = lane >> 4;
  const int wm   = (wid & 3) * 32;   // wave M offset within block tile
  const int wn   = (wid >> 2) * 64;  // wave N offset within block tile
  const long m0  = (long)blockIdx.y * 128;
  const long n0  = (long)blockIdx.x * 128;

  f8_t acc[2][4] = {};

  for (int k0 = 0; k0 < K; k0 += 32) {
    // ---- stage A tile (128 x 32) into LDS as f16 ----
    {
      const int r = tid >> 1;
      const int c = (tid & 1) * 16;
      if (A16) {
        const _Float16* A = (const _Float16*)Ap;
        const h8_t* src = (const h8_t*)(A + (m0 + r) * (long)K + k0 + c);
        h8_t x0 = src[0], x1 = src[1];
        *(h8_t*)&As[r][c]     = x0;
        *(h8_t*)&As[r][c + 8] = x1;
      } else {
        const float* A = (const float*)Ap;
        const float4* src = (const float4*)(A + (m0 + r) * (long)K + k0 + c);
        float4 f0 = src[0], f1 = src[1], f2 = src[2], f3 = src[3];
        *(h8_t*)&As[r][c]     = cvt8(f0, f1);
        *(h8_t*)&As[r][c + 8] = cvt8(f2, f3);
      }
    }
    // ---- stage B tile (32 x 128) transposed into LDS as f16 ----
    {
      const int kr = tid >> 3;
      const int nc = (tid & 7) * 16;
      const float4* src = (const float4*)(Bp + (long)(k0 + kr) * N + n0 + nc);
      float4 f0 = src[0], f1 = src[1], f2 = src[2], f3 = src[3];
      float vals[16] = { f0.x, f0.y, f0.z, f0.w, f1.x, f1.y, f1.z, f1.w,
                         f2.x, f2.y, f2.z, f2.w, f3.x, f3.y, f3.z, f3.w };
#pragma unroll
      for (int j = 0; j < 16; ++j) Bt[nc + j][kr] = (_Float16)vals[j];
    }
    __syncthreads();

    // ---- fragments + 8 WMMAs ----
    h16_t afr[2], bfr[4];
#pragma unroll
    for (int i = 0; i < 2; ++i) {
      const _Float16* p = &As[wm + i * 16 + cl][0];
      afr[i] = cat8(*(const h8_t*)(p + hi * 8), *(const h8_t*)(p + 16 + hi * 8));
    }
#pragma unroll
    for (int j = 0; j < 4; ++j) {
      const _Float16* p = &Bt[wn + j * 16 + cl][0];
      bfr[j] = cat8(*(const h8_t*)(p + hi * 8), *(const h8_t*)(p + 16 + hi * 8));
    }
#pragma unroll
    for (int i = 0; i < 2; ++i)
#pragma unroll
      for (int j = 0; j < 4; ++j)
        acc[i][j] = wmma16(afr[i], bfr[j], acc[i][j]);
    __syncthreads();
  }

  // ---- epilogue: C/D layout row = v + 8*hi, col = cl ----
#pragma unroll
  for (int i = 0; i < 2; ++i)
#pragma unroll
    for (int j = 0; j < 4; ++j)
#pragma unroll
      for (int v = 0; v < 8; ++v) {
        const long row = m0 + wm + i * 16 + v + 8 * hi;
        const long col = n0 + wn + j * 16 + cl;
        if (OUT32) ((float*)Cp)[row * N + col] = acc[i][j][v];
        else       ((_Float16*)Cp)[row * N + col] = (_Float16)acc[i][j][v];
      }
}

// ---------------------------------------------------------------------------
// RoPE (half-split rotation) applied in place to f16 Q or K: x[b][s][nh][256]
// ---------------------------------------------------------------------------
__global__ __launch_bounds__(256) void rope_kernel(
    _Float16* __restrict__ x, const float* __restrict__ cosb,
    const float* __restrict__ sinb, int nheads)
{
  const long i  = (long)blockIdx.x * blockDim.x + threadIdx.x;
  const int  d  = (int)(i & 127);
  const long t  = i >> 7;          // (b*S + s)*nheads + h
  const long bs = t / nheads;      // b*S + s
  const long base = t * D_;
  const float c  = cosb[bs * 128 + d];
  const float sn = sinb[bs * 128 + d];
  const float x1 = (float)x[base + d];
  const float x2 = (float)x[base + 128 + d];
  x[base + d]       = (_Float16)(x1 * c - x2 * sn);
  x[base + 128 + d] = (_Float16)(x1 * sn + x2 * c);
}

// ---------------------------------------------------------------------------
// V transpose: v[b][s][kh][d] -> vt[b][kh][d][s] (contiguous along keys)
// ---------------------------------------------------------------------------
__global__ __launch_bounds__(256) void transpose_v_kernel(
    const _Float16* __restrict__ v, _Float16* __restrict__ vt)
{
  const long i  = (long)blockIdx.x * blockDim.x + threadIdx.x; // B*S*HKV*D
  const int  d  = (int)(i & 255);
  const long t  = i >> 8;          // (b*S + s)*HKV + kh
  const int  kh = (int)(t & 7);
  const long bs = t >> 3;          // b*S + s
  const long s  = bs & (S_ - 1);
  const long b  = bs >> 11;
  vt[((b * HKV_ + kh) * (long)D_ + d) * S_ + s] = v[i];
}

// ---------------------------------------------------------------------------
// Flash attention (GQA, causal): 128 threads = 4 waves; 64 q-rows per WG;
// key blocks of 32; per wave: 16 q-rows, full D=256 accumulator (16 tiles).
// ---------------------------------------------------------------------------
__global__ __launch_bounds__(128) void attn_kernel(
    const _Float16* __restrict__ q16, const _Float16* __restrict__ k16,
    const _Float16* __restrict__ vt16, _Float16* __restrict__ attn)
{
  __shared__ _Float16 Qs[64][264];     // 33.8 KB  (264 halves = 528B, 16B-aligned)
  __shared__ _Float16 Ks[32][264];     // 16.9 KB
  __shared__ _Float16 Ps[4][16][40];   //  5.1 KB  per-wave P tiles

  const int tid  = threadIdx.x;
  const int lane = tid & 31;
  const int wid  = tid >> 5;
  const int cl   = lane & 15;
  const int hi   = lane >> 4;
  const int q0   = blockIdx.x * 64;
  const int b    = blockIdx.y >> 4;    // blockIdx.y = b*H + h
  const int h    = blockIdx.y & 15;
  const int kh   = h >> 1;             // G = H/HKV = 2

  // ---- stage Q tile (64 rows x 256) once ----
  for (int c = tid; c < 64 * 32; c += 128) {
    const int r  = c >> 5;
    const int c8 = (c & 31) * 8;
    const h8_t* src =
        (const h8_t*)(q16 + (((long)(b * S_ + q0 + r) * H_ + h) * D_ + c8));
    *(h8_t*)&Qs[r][c8] = *src;
  }
  __syncthreads();

  f8_t out[16] = {};
  float rmax[8], rsum[8];
#pragma unroll
  for (int v = 0; v < 8; ++v) { rmax[v] = -3.0e38f; rsum[v] = 0.0f; }

  const int nblk = q0 / 32 + 2;        // causal: keys <= q0+63
  for (int blk = 0; blk < nblk; ++blk) {
    const int kb = blk * 32;

    // ---- stage K block (32 keys x 256) ----
    for (int c = tid; c < 32 * 32; c += 128) {
      const int r  = c >> 5;
      const int c8 = (c & 31) * 8;
      const h8_t* src =
          (const h8_t*)(k16 + (((long)(b * S_ + kb + r) * HKV_ + kh) * D_ + c8));
      *(h8_t*)&Ks[r][c8] = *src;
    }
    __syncthreads();

    // ---- scores: S = Q * K^T over D=256 (8 k-steps x 2 n-tiles) ----
    f8_t sc[2] = {};
#pragma unroll
    for (int ks = 0; ks < 8; ++ks) {
      const _Float16* qp = &Qs[wid * 16 + cl][ks * 32];
      h16_t a = cat8(*(const h8_t*)(qp + hi * 8),
                     *(const h8_t*)(qp + 16 + hi * 8));
#pragma unroll
      for (int nt = 0; nt < 2; ++nt) {
        const _Float16* kp = &Ks[nt * 16 + cl][ks * 32];
        h16_t bb = cat8(*(const h8_t*)(kp + hi * 8),
                        *(const h8_t*)(kp + 16 + hi * 8));
        sc[nt] = wmma16(a, bb, sc[nt]);
      }
    }

    // ---- scale + causal mask ----
    const int rowg = q0 + wid * 16 + 8 * hi;   // + v
#pragma unroll
    for (int nt = 0; nt < 2; ++nt) {
      const int key = kb + nt * 16 + cl;
#pragma unroll
      for (int v = 0; v < 8; ++v) {
        const float s = sc[nt][v] * 0.0625f;   // D^-0.5 = 1/16
        sc[nt][v] = (key <= rowg + v) ? s : -1.0e9f;
      }
    }

    // ---- online softmax (row reductions across the 16-lane half) ----
    float e0[8], e1[8];
#pragma unroll
    for (int v = 0; v < 8; ++v) {
      float m = fmaxf(sc[0][v], sc[1][v]);
#pragma unroll
      for (int off = 8; off >= 1; off >>= 1)
        m = fmaxf(m, __shfl_xor(m, off, 16));
      const float mnew = fmaxf(rmax[v], m);
      const float f    = __expf(rmax[v] - mnew);
      const float p0   = __expf(sc[0][v] - mnew);
      const float p1   = __expf(sc[1][v] - mnew);
      float ssum = p0 + p1;
#pragma unroll
      for (int off = 8; off >= 1; off >>= 1)
        ssum += __shfl_xor(ssum, off, 16);
      rsum[v] = rsum[v] * f + ssum;
      rmax[v] = mnew;
      e0[v] = p0; e1[v] = p1;
#pragma unroll
      for (int t = 0; t < 16; ++t) out[t][v] *= f;   // rescale accumulator
    }

    // ---- P -> LDS (C-layout -> A-layout transpose; same-wave DS in-order) ----
#pragma unroll
    for (int v = 0; v < 8; ++v) {
      Ps[wid][8 * hi + v][cl]      = (_Float16)e0[v];
      Ps[wid][8 * hi + v][16 + cl] = (_Float16)e1[v];
    }

    // ---- O += P * V (A = P 16x32, B = Vt rows contiguous over keys) ----
    {
      const _Float16* pp = &Ps[wid][cl][0];
      h16_t a = cat8(*(const h8_t*)(pp + hi * 16),
                     *(const h8_t*)(pp + hi * 16 + 8));
#pragma unroll
      for (int t = 0; t < 16; ++t) {
        const _Float16* vp =
            vt16 + (((long)(b * HKV_ + kh) * D_ + t * 16 + cl) * S_ + kb + hi * 16);
        h16_t bb = cat8(*(const h8_t*)vp, *(const h8_t*)(vp + 8));
        out[t] = wmma16(a, bb, out[t]);
      }
    }
    __syncthreads();
  }

  // ---- epilogue: normalize and store attn[b][s][h][d] as f16 ----
#pragma unroll
  for (int t = 0; t < 16; ++t)
#pragma unroll
    for (int v = 0; v < 8; ++v) {
      const long row = q0 + wid * 16 + 8 * hi + v;
      const long col = t * 16 + cl;
      attn[(((long)b * S_ + row) * H_ + h) * D_ + col] =
          (_Float16)(out[t][v] / rsum[v]);
    }
}

// ---------------------------------------------------------------------------
// Launch
// ---------------------------------------------------------------------------
extern "C" void kernel_launch(void* const* d_in, const int* in_sizes, int n_in,
                              void* d_out, int out_size, void* d_ws, size_t ws_size,
                              hipStream_t stream)
{
  (void)in_sizes; (void)n_in; (void)out_size;

  const float* hidden = (const float*)d_in[0];
  const float* cosb   = (const float*)d_in[1];
  const float* sinb   = (const float*)d_in[2];
  // d_in[3] = mask (causal mask applied analytically in-kernel)
  const float* wq = (const float*)d_in[4];
  const float* wk = (const float*)d_in[5];
  const float* wv = (const float*)d_in[6];
  const float* wo = (const float*)d_in[7];
  float* out = (float*)d_out;

  // workspace layout (bytes): 112 MB total
  char* ws = (char*)d_ws;
  const size_t QB = (size_t)B_ * S_ * H_   * D_ * 2;  // 33,554,432
  const size_t KB = (size_t)B_ * S_ * HKV_ * D_ * 2;  // 16,777,216
  if (ws_size < QB + 3 * KB + QB) return;
  _Float16* q16    = (_Float16*)(ws);
  _Float16* k16    = (_Float16*)(ws + QB);
  _Float16* v16    = (_Float16*)(ws + QB + KB);
  _Float16* vt16   = (_Float16*)(ws + QB + 2 * KB);
  _Float16* attn16 = (_Float16*)(ws + QB + 3 * KB);

  const int M = B_ * S_;                 // 4096
  const dim3 blk256(256);

  // 1) QKV projections (fp32 in, f16 out)
  gemm_wmma_kernel<false, false>
      <<<dim3(H_ * D_ / 128, M / 128), blk256, 0, stream>>>(hidden, wq, q16, M, H_ * D_, HID_);
  gemm_wmma_kernel<false, false>
      <<<dim3(HKV_ * D_ / 128, M / 128), blk256, 0, stream>>>(hidden, wk, k16, M, HKV_ * D_, HID_);
  gemm_wmma_kernel<false, false>
      <<<dim3(HKV_ * D_ / 128, M / 128), blk256, 0, stream>>>(hidden, wv, v16, M, HKV_ * D_, HID_);

  // 2) RoPE on Q and K; transpose V
  rope_kernel<<<(B_ * S_ * H_   * 128) / 256, 256, 0, stream>>>(q16, cosb, sinb, H_);
  rope_kernel<<<(B_ * S_ * HKV_ * 128) / 256, 256, 0, stream>>>(k16, cosb, sinb, HKV_);
  transpose_v_kernel<<<(B_ * S_ * HKV_ * D_) / 256, 256, 0, stream>>>(v16, vt16);

  // 3) causal GQA flash attention
  attn_kernel<<<dim3(S_ / 64, B_ * H_), dim3(128), 0, stream>>>(q16, k16, vt16, attn16);

  // 4) output projection (f16 in, fp32 out)
  gemm_wmma_kernel<true, true>
      <<<dim3(HID_ / 128, M / 128), blk256, 0, stream>>>(attn16, wo, out, M, HID_, H_ * D_);
}
